// GPTNeoX_36189394436340
// MI455X (gfx1250) — matrix operand order repaired and checked by reading, not compiled, not run on hardware
//
#include <hip/hip_runtime.h>
#include <hip/hip_bf16.h>

// ---------------------------------------------------------------------------
// Model constants (match reference)
// ---------------------------------------------------------------------------
#define NTOK   2048
#define DMODEL 1024
#define NHEAD  16
#define DHEAD  64
#define INNER  1024           // NHEAD*DHEAD
#define QKVW   3072           // 3*INNER
#define FFDIM  4096
#define NDEPTH 2
#define VOCAB  50257
#define LN_EPS 1e-5f

// gfx1250 async global->LDS staging (ASYNCcnt / s_wait_asynccnt path).
// Fallback (0): plain load + ds_store.
#define USE_ASYNC_LDS 1

typedef __attribute__((ext_vector_type(8)))  float  v8f;
typedef __attribute__((ext_vector_type(16))) __bf16 v16bf;
typedef unsigned short u16;

// ---------------------------------------------------------------------------
// Helpers
// ---------------------------------------------------------------------------
__device__ __forceinline__ u16 f2bf(float f) {
    unsigned int u = __float_as_uint(f);
    unsigned int r = u + 0x7FFFu + ((u >> 16) & 1u);   // round-to-nearest-even
    return (u16)(r >> 16);
}

// Build a 16-element bf16 fragment (32 bytes = 8 VGPRs) from two 16B chunks.
__device__ __forceinline__ v16bf ldfrag(const u16* p0, const u16* p1) {
    union { uint4 q[2]; v16bf v; } f;
    f.q[0] = *reinterpret_cast<const uint4*>(p0);
    f.q[1] = *reinterpret_cast<const uint4*>(p1);
    return f.v;
}

__device__ __forceinline__ v8f wmma_bf16(v16bf a, v16bf b, v8f c) {
    // D(f32 16x16) = A(16x32 bf16) * B(32x16 bf16) + C
    return __builtin_amdgcn_wmma_f32_16x16x32_bf16(
        false, a, false, b, (short)0, c, false, false);
}

// A-fragment (16x32, row-major, ld in elems): row = lane&15, half = lane>>4
__device__ __forceinline__ v16bf ldfragA(const u16* row, int k0, int half) {
    return ldfrag(row + k0 + half * 8, row + k0 + 16 + half * 8);
}
// B-fragment (32x16) from K-contiguous storage: col = lane&15, K contiguous
__device__ __forceinline__ v16bf ldfragB(const u16* col, int k0, int half) {
    return ldfrag(col + k0 + half * 8, col + k0 + 16 + half * 8);
}

// 16B global -> LDS copy (async on CDNA5)
#if USE_ASYNC_LDS
#define CP16(lptr, gptr)                                                      \
    asm volatile("global_load_async_to_lds_b128 %0, %1, off"                  \
                 :: "v"((unsigned)(size_t)(lptr)),                            \
                    "v"((unsigned long long)(size_t)(gptr)) : "memory")
#define WAIT_ASYNC_LE5() asm volatile("s_wait_asynccnt 0x5" ::: "memory")
#define WAIT_ASYNC_0()   asm volatile("s_wait_asynccnt 0x0" ::: "memory")
#else
#define CP16(lptr, gptr)                                                      \
    (*reinterpret_cast<uint4*>(lptr) = *reinterpret_cast<const uint4*>(gptr))
#define WAIT_ASYNC_LE5()
#define WAIT_ASYNC_0()
#endif

// ---------------------------------------------------------------------------
// Embedding: h[n,d] = tok_emb[x[n],d] + pos_emb[n,d]   (f32)
// ---------------------------------------------------------------------------
__global__ void embed_kernel(const int* __restrict__ x,
                             const float* __restrict__ tok,
                             const float* __restrict__ pos,
                             float* __restrict__ h) {
    int i = blockIdx.x * 256 + threadIdx.x;
    if (i >= NTOK * DMODEL) return;
    int n = i / DMODEL, d = i - n * DMODEL;
    h[i] = tok[(size_t)x[n] * DMODEL + d] + pos[i];
}

// f32 -> bf16 flat convert
__global__ void cvt_bf16_kernel(const float* __restrict__ in,
                                u16* __restrict__ out, int count) {
    int i = blockIdx.x * 256 + threadIdx.x;
    if (i < count) out[i] = f2bf(in[i]);
}

// f32 [R,C] -> bf16 [C,R]
__global__ void transpose_bf16_kernel(const float* __restrict__ in,
                                      u16* __restrict__ out, int R, int C) {
    int i = blockIdx.x * 256 + threadIdx.x;
    if (i >= R * C) return;
    int r = i / C, c = i - r * C;
    out[(size_t)c * R + r] = f2bf(in[i]);
}

// ---------------------------------------------------------------------------
// LayerNorm over last dim -> bf16 activations
// ---------------------------------------------------------------------------
__global__ __launch_bounds__(256)
void ln_kernel(const float* __restrict__ X, const float* __restrict__ g,
               const float* __restrict__ b, u16* __restrict__ Y, int Dm) {
    __shared__ float red[256];
    const int row = blockIdx.x, tid = threadIdx.x;
    const float* xr = X + (size_t)row * Dm;

    float s = 0.f;
    for (int i = tid; i < Dm; i += 256) s += xr[i];
    red[tid] = s; __syncthreads();
    for (int w = 128; w > 0; w >>= 1) {
        if (tid < w) red[tid] += red[tid + w];
        __syncthreads();
    }
    const float mu = red[0] / (float)Dm;
    __syncthreads();

    float v = 0.f;
    for (int i = tid; i < Dm; i += 256) { float d = xr[i] - mu; v += d * d; }
    red[tid] = v; __syncthreads();
    for (int w = 128; w > 0; w >>= 1) {
        if (tid < w) red[tid] += red[tid + w];
        __syncthreads();
    }
    const float rstd = rsqrtf(red[0] / (float)Dm + LN_EPS);

    u16* yr = Y + (size_t)row * Dm;
    for (int i = tid; i < Dm; i += 256)
        yr[i] = f2bf((xr[i] - mu) * rstd * g[i] + b[i]);
}

// ---------------------------------------------------------------------------
// bf16 WMMA GEMM, LDS-staged with double-buffered async global->LDS copies.
//   C[M,Nout] = A[M,K](bf16 row-major) * Bt[Nout,K](bf16 K-contiguous)
// Block: 256 threads = 8 waves; block tile 64(M) x 256(N), BK=32.
// Wave tile: 32x64 (2 M-subtiles x 4 N-subtiles = 8 WMMAs per K-step).
//   MODE 0: Cf[m,n] += acc + bias[n]          (f32 residual, in-place)
//   MODE 1: Cb[m,n]  = bf16(gelu(acc+bias))
//   MODE 2: Cb[m,n]  = bf16(acc)              (no bias)
//   MODE 3: Cf[m,n]  = acc (bounds-checked n) (logits)
// ---------------------------------------------------------------------------
template <int MODE>
__global__ __launch_bounds__(256)
void gemm_bf16_kernel(const u16* __restrict__ A, int lda,
                      const u16* __restrict__ Bt, int ldb,
                      const float* __restrict__ bias,
                      float* __restrict__ Cf, u16* __restrict__ Cb,
                      int M, int Nout, int K) {
    __shared__ __align__(16) u16 sA[2][64 * 32];    // 4KB  x2
    __shared__ __align__(16) u16 sB[2][256 * 32];   // 16KB x2

    const int tid  = threadIdx.x;
    const int wave = tid >> 5;
    const int lane = tid & 31;
    const int half = lane >> 4;
    const int l16  = lane & 15;
    const int wm   = wave >> 2;          // 0..1  (M)
    const int wn   = wave & 3;           // 0..3  (N)

    const int bm = blockIdx.y * 64;
    const int bn = blockIdx.x * 256;

    // --- stage one BK=32 slab of A (64x32) and B (256x32) into LDS buf ---
    auto stage = [&](int buf, int kk) {
        {   // A: 256 x 16B chunks, one per thread
            const int row = tid >> 2, c4 = (tid & 3) * 8;
            const u16* g = A + (size_t)(bm + row) * lda + kk + c4;
            CP16(&sA[buf][row * 32 + c4], g);
        }
#pragma unroll
        for (int i = 0; i < 4; ++i) {   // B: 1024 chunks, 4 per thread
            const int id = tid + i * 256;
            const int col = id >> 2, c4 = (id & 3) * 8;
            int nc = bn + col; if (nc > Nout - 1) nc = Nout - 1;
            const u16* g = Bt + (size_t)nc * ldb + kk + c4;
            CP16(&sB[buf][col * 32 + c4], g);
        }
    };

    v8f acc[2][4] = {};

    stage(0, 0);
    for (int kk = 0; kk < K; kk += 32) {
        const int cur = (kk >> 5) & 1;
        if (kk + 32 < K) {
            stage(cur ^ 1, kk + 32);    // prefetch next slab
            WAIT_ASYNC_LE5();           // wait: current slab's 5 copies done
        } else {
            WAIT_ASYNC_0();
        }
        __syncthreads();                // all waves' writes to cur visible

        // A fragments (reused across the 4 N-subtiles)
        v16bf af[2];
#pragma unroll
        for (int mt = 0; mt < 2; ++mt) {
            const u16* ar = &sA[cur][(wm * 32 + mt * 16 + l16) * 32];
            af[mt] = ldfragA(ar, 0, half);
        }
#pragma unroll
        for (int nt = 0; nt < 4; ++nt) {
            const u16* bc = &sB[cur][(wn * 64 + nt * 16 + l16) * 32];
            v16bf bf = ldfragB(bc, 0, half);
            acc[0][nt] = wmma_bf16(af[0], bf, acc[0][nt]);
            acc[1][nt] = wmma_bf16(af[1], bf, acc[1][nt]);
        }
        __syncthreads();                // reads of cur done before overwrite
    }

    // ---- epilogue ----
    const int n_base = bn + wn * 64;
    const int m_base = bm + wm * 32;
#pragma unroll
    for (int mt = 0; mt < 2; ++mt) {
#pragma unroll
        for (int nt = 0; nt < 4; ++nt) {
            const int nc = n_base + nt * 16 + l16;
            float bi = 0.f;
            if (MODE == 0 || MODE == 1) bi = bias[nc];
            v8f a = acc[mt][nt];
#pragma unroll
            for (int r = 0; r < 8; ++r) {
                const int m = m_base + mt * 16 + r + half * 8;
                const size_t idx = (size_t)m * Nout + nc;
                float v = a[r] + bi;
                if (MODE == 0) {
                    Cf[idx] = Cf[idx] + v;
                } else if (MODE == 1) {
                    float gl = 0.5f * v * (1.0f + erff(v * 0.70710678118654752f));
                    Cb[idx] = f2bf(gl);
                } else if (MODE == 2) {
                    Cb[idx] = f2bf(a[r]);
                } else {
                    if (nc < Nout) Cf[idx] = a[r];
                }
            }
        }
    }
}

// ---------------------------------------------------------------------------
// Causal flash attention. One wave per (head, 16-query tile).
// qkv: bf16 [NTOK, 3*INNER]; o: bf16 [NTOK, INNER]
// ---------------------------------------------------------------------------
__global__ __launch_bounds__(32)
void attn_kernel(const u16* __restrict__ qkv, u16* __restrict__ o, float scale) {
    __shared__ float sS[16][33];
    __shared__ __align__(16) u16 sP[16][32];
    __shared__ __align__(16) u16 sVT[64][32];
    __shared__ float sM[16], sL[16], sC[16], sInv[16];

    const int lane = threadIdx.x;
    const int l16  = lane & 15;
    const int half = lane >> 4;
    const int hd   = blockIdx.x;
    const int m0   = blockIdx.y * 16;

    const u16* qbase = qkv + (size_t)(m0 + l16) * QKVW + hd * DHEAD;
    const v16bf aQ0 = ldfragA(qbase, 0, half);
    const v16bf aQ1 = ldfragA(qbase, 32, half);

    v8f accO[4] = {};
    if (lane < 16) { sM[lane] = -3.0e38f; sL[lane] = 0.f; }
    __syncthreads();

    const int kmax = m0 + 15;
    for (int j0 = 0; j0 <= kmax; j0 += 32) {
        // hoist V row loads (key = j0+lane) so they overlap S + softmax
        const u16* vbase = qkv + (size_t)(j0 + lane) * QKVW + 2 * INNER + hd * DHEAD;
        union { uint4 q[8]; u16 u[64]; } vv;
#pragma unroll
        for (int i = 0; i < 8; ++i)
            vv.q[i] = reinterpret_cast<const uint4*>(vbase)[i];

        // ---- S = scale * Q K^T (all 4 K fragments batched, then 4 WMMAs) ----
        const u16* kb0 = qkv + (size_t)(j0 + l16) * QKVW + INNER + hd * DHEAD;
        const u16* kb1 = qkv + (size_t)(j0 + 16 + l16) * QKVW + INNER + hd * DHEAD;
        v16bf b00 = ldfragB(kb0, 0, half), b01 = ldfragB(kb0, 32, half);
        v16bf b10 = ldfragB(kb1, 0, half), b11 = ldfragB(kb1, 32, half);
        v8f s0 = {}, s1 = {};
        s0 = wmma_bf16(aQ0, b00, s0);
        s0 = wmma_bf16(aQ1, b01, s0);
        s1 = wmma_bf16(aQ0, b10, s1);
        s1 = wmma_bf16(aQ1, b11, s1);

        const int key0 = j0 + l16, key1 = j0 + 16 + l16;
#pragma unroll
        for (int r = 0; r < 8; ++r) {
            const int mr = r + half * 8;
            const int qrow = m0 + mr;
            sS[mr][l16]      = (key0 <= qrow) ? s0[r] * scale : -3.0e38f;
            sS[mr][16 + l16] = (key1 <= qrow) ? s1[r] * scale : -3.0e38f;
        }
        __syncthreads();

        // ---- online softmax (lanes 0..15 own one row each) ----
        if (lane < 16) {
            float mold = sM[lane];
            float mx = mold;
            for (int c = 0; c < 32; ++c) mx = fmaxf(mx, sS[lane][c]);
            float corr = __expf(mold - mx);
            float sum = 0.f;
            for (int c = 0; c < 32; ++c) {
                float p = __expf(sS[lane][c] - mx);
                sum += p;
                sP[lane][c] = f2bf(p);
            }
            sL[lane] = sL[lane] * corr + sum;
            sM[lane] = mx;
            sC[lane] = corr;
        }
        __syncthreads();

        // rescale accumulators
        float cr[8];
#pragma unroll
        for (int r = 0; r < 8; ++r) cr[r] = sC[r + half * 8];
#pragma unroll
        for (int t = 0; t < 4; ++t)
#pragma unroll
            for (int r = 0; r < 8; ++r) accO[t][r] *= cr[r];

        // scatter V^T into LDS (already fetched)
#pragma unroll
        for (int d = 0; d < 64; ++d) sVT[d][lane] = vv.u[d];
        __syncthreads();

        // ---- O += P(16x32) @ V(32x64) ----
        v16bf aP = ldfrag(&sP[l16][half * 8], &sP[l16][16 + half * 8]);
#pragma unroll
        for (int t = 0; t < 4; ++t) {
            v16bf bv = ldfrag(&sVT[t * 16 + l16][half * 8],
                              &sVT[t * 16 + l16][16 + half * 8]);
            accO[t] = wmma_bf16(aP, bv, accO[t]);
        }
        __syncthreads();
    }

    if (lane < 16) sInv[lane] = 1.0f / sL[lane];
    __syncthreads();
    float ir[8];
#pragma unroll
    for (int r = 0; r < 8; ++r) ir[r] = sInv[r + half * 8];

#pragma unroll
    for (int t = 0; t < 4; ++t)
#pragma unroll
        for (int r = 0; r < 8; ++r) {
            const int m = m0 + r + half * 8;
            const int col = hd * DHEAD + t * 16 + l16;
            o[(size_t)m * INNER + col] = f2bf(accO[t][r] * ir[r]);
        }
}

// ---------------------------------------------------------------------------
// Host-side orchestration
// ---------------------------------------------------------------------------
extern "C" void kernel_launch(void* const* d_in, const int* in_sizes, int n_in,
                              void* d_out, int out_size, void* d_ws, size_t ws_size,
                              hipStream_t stream) {
    (void)in_sizes; (void)n_in; (void)out_size; (void)ws_size;

    const int*   x     = (const int*)  d_in[0];
    const float* tok   = (const float*)d_in[1];
    const float* pos   = (const float*)d_in[2];
    const float* ln1_g = (const float*)d_in[3];
    const float* ln1_b = (const float*)d_in[4];
    const float* wqkv  = (const float*)d_in[5];
    const float* wo    = (const float*)d_in[6];
    const float* bo    = (const float*)d_in[7];
    const float* ln2_g = (const float*)d_in[8];
    const float* ln2_b = (const float*)d_in[9];
    const float* w1    = (const float*)d_in[10];
    const float* b1    = (const float*)d_in[11];
    const float* w2    = (const float*)d_in[12];
    const float* b2    = (const float*)d_in[13];
    const float* lnf_g = (const float*)d_in[14];
    const float* lnf_b = (const float*)d_in[15];

    char* ws = (char*)d_ws;
    size_t off = 0;
    auto carve = [&](size_t bytes) -> char* {
        char* p = ws + off;
        off = (off + bytes + 255) & ~(size_t)255;
        return p;
    };
    float* hbuf   = (float*)carve((size_t)NTOK * DMODEL * 4);
    u16*   ybuf   = (u16*)  carve((size_t)NTOK * DMODEL * 2);
    u16*   qkvbuf = (u16*)  carve((size_t)NTOK * QKVW * 2);
    u16*   attbuf = (u16*)  carve((size_t)NTOK * INNER * 2);
    u16*   ffbuf  = (u16*)  carve((size_t)NTOK * FFDIM * 2);
    u16*   wqkvT  = (u16*)  carve((size_t)QKVW * DMODEL * 2);
    u16*   woT    = (u16*)  carve((size_t)DMODEL * INNER * 2);
    u16*   w1T    = (u16*)  carve((size_t)FFDIM * DMODEL * 2);
    u16*   w2T    = (u16*)  carve((size_t)DMODEL * FFDIM * 2);
    u16*   tokb   = (u16*)  carve((size_t)VOCAB * DMODEL * 2);

    auto g1 = [](int n) { return (n + 255) / 256; };

    embed_kernel<<<g1(NTOK * DMODEL), 256, 0, stream>>>(x, tok, pos, hbuf);
    cvt_bf16_kernel<<<g1(VOCAB * DMODEL), 256, 0, stream>>>(tok, tokb, VOCAB * DMODEL);

    const float scale = 0.125f;   // DHEAD^-0.5

    for (int l = 0; l < NDEPTH; ++l) {
        const float* wqkv_l = wqkv + (size_t)l * DMODEL * QKVW;
        const float* wo_l   = wo   + (size_t)l * INNER * DMODEL;
        const float* w1_l   = w1   + (size_t)l * DMODEL * FFDIM;
        const float* w2_l   = w2   + (size_t)l * FFDIM * DMODEL;

        transpose_bf16_kernel<<<g1(DMODEL * QKVW), 256, 0, stream>>>(wqkv_l, wqkvT, DMODEL, QKVW);
        transpose_bf16_kernel<<<g1(INNER * DMODEL), 256, 0, stream>>>(wo_l, woT, INNER, DMODEL);
        transpose_bf16_kernel<<<g1(DMODEL * FFDIM), 256, 0, stream>>>(w1_l, w1T, DMODEL, FFDIM);
        transpose_bf16_kernel<<<g1(FFDIM * DMODEL), 256, 0, stream>>>(w2_l, w2T, FFDIM, DMODEL);

        // --- attention block ---
        ln_kernel<<<NTOK, 256, 0, stream>>>(hbuf, ln1_g + l * DMODEL, ln1_b + l * DMODEL, ybuf, DMODEL);
        gemm_bf16_kernel<2><<<dim3(QKVW / 256, NTOK / 64), 256, 0, stream>>>(
            ybuf, DMODEL, wqkvT, DMODEL, nullptr, nullptr, qkvbuf, NTOK, QKVW, DMODEL);
        attn_kernel<<<dim3(NHEAD, NTOK / 16), 32, 0, stream>>>(qkvbuf, attbuf, scale);
        gemm_bf16_kernel<0><<<dim3(DMODEL / 256, NTOK / 64), 256, 0, stream>>>(
            attbuf, INNER, woT, INNER, bo + l * DMODEL, hbuf, nullptr, NTOK, DMODEL, INNER);

        // --- feed-forward block ---
        ln_kernel<<<NTOK, 256, 0, stream>>>(hbuf, ln2_g + l * DMODEL, ln2_b + l * DMODEL, ybuf, DMODEL);
        gemm_bf16_kernel<1><<<dim3(FFDIM / 256, NTOK / 64), 256, 0, stream>>>(
            ybuf, DMODEL, w1T, DMODEL, b1 + l * FFDIM, nullptr, ffbuf, NTOK, FFDIM, DMODEL);
        gemm_bf16_kernel<0><<<dim3(DMODEL / 256, NTOK / 64), 256, 0, stream>>>(
            ffbuf, FFDIM, w2T, FFDIM, b2 + l * DMODEL, hbuf, nullptr, NTOK, DMODEL, FFDIM);
    }

    ln_kernel<<<NTOK, 256, 0, stream>>>(hbuf, lnf_g, lnf_b, ybuf, DMODEL);
    gemm_bf16_kernel<3><<<dim3((VOCAB + 255) / 256, NTOK / 64), 256, 0, stream>>>(
        ybuf, DMODEL, tokb, DMODEL, nullptr, (float*)d_out, nullptr, NTOK, VOCAB, DMODEL);
}
